// PDIAE_Encoder_16149077033459
// MI455X (gfx1250) — matrix operand order repaired and checked by reading, not compiled, not run on hardware
//
#include <hip/hip_runtime.h>
#include <hip/hip_bf16.h>

// ---------------------------------------------------------------------------
// out[b,c,k1,k2] = f1(k1)*X1[k1,k2] + f2(k2)*X2[k1,k2]   (complex64)
//   X1 = rfft2(x * i/255)/256,  X2 = rfft2(x * j/255)/256 (ortho)
//   nonzero only for k1<64, k2 in [0,64) U [65,129).
// Factored as WMMA matmuls:
//   Stage1: U=x*Bc, V=x*Bs, Uw=x*Bwc, Vw=x*Bws           (256x256 x 256x512)
//   Stage2: re ~ f1*(Awc*U - Aws*V) + f2*(Ac*Uw - As*Vw) (64x256 x 256x128)
//           im ~ -[f1*(Awc*V + Aws*U) + f2*(Ac*Vw + As*Uw)]
// f16 inputs, f32 accumulate via v_wmma_f32_16x16x32_f16 (wave32).
// ---------------------------------------------------------------------------

typedef __attribute__((ext_vector_type(16))) _Float16 v16h;
typedef __attribute__((ext_vector_type(8)))  float    v8f;

#define S1 256
#define KCOLS 128          // kept k2 columns (0..63 -> k2, 64..127 -> k2+1)
#define M1 64              // kept k1 rows
#define LDSTR 264          // LDS row stride in halves (256 + 8 pad)
#define NB_HALVES (512 * 256)       // stage-1 basis Bt[col][j]
#define NA_HALVES (6 * 64 * 256)    // stage-2 basis A8[var][k1][i]
#define TWO_PI_OVER_256 0.024543692606170259f

// ---------------- basis init (deterministic, rebuilt each call) ------------
__global__ void basis_init_kernel(_Float16* __restrict__ bt,
                                  _Float16* __restrict__ a8) {
  int idx = blockIdx.x * blockDim.x + threadIdx.x;
  if (idx < NB_HALVES) {
    int col = idx >> 8;          // 0..511
    int j   = idx & 255;
    int v   = col >> 7;          // 0:cos 1:sin 2:w*cos 3:w*sin
    int k2c = col & 127;
    int k2  = (k2c < 64) ? k2c : k2c + 1;
    int t   = (k2 * j) & 255;    // exact angle reduction (period 256)
    float th = t * TWO_PI_OVER_256;
    float val = (v & 1) ? __sinf(th) : __cosf(th);
    if (v >= 2) val *= j * (1.0f / 255.0f);
    bt[idx] = (_Float16)val;
  } else {
    int r = idx - NB_HALVES;
    if (r < NA_HALVES) {
      int var = r >> 14;         // 0..5
      int rem = r & 16383;
      int k1  = rem >> 8;
      int i   = rem & 255;
      int t   = (k1 * i) & 255;
      float th = t * TWO_PI_OVER_256;
      float c = __cosf(th), s = __sinf(th);
      float w = i * (1.0f / 255.0f);
      float val;
      switch (var) {
        case 0: val =  w * c; break;   // Awc
        case 1: val =  w * s; break;   // Aws
        case 2: val = -w * s; break;   // -Aws
        case 3: val =  c;     break;   // Ac
        case 4: val =  s;     break;   // As
        default: val = -s;    break;   // -As
      }
      a8[r] = (_Float16)val;
    }
  }
}

// ---------------- zero-fill output (rows >=64 and col 64 stay zero) --------
__global__ void zero_out_kernel(float4* __restrict__ p, long n4) {
  long i = (long)blockIdx.x * blockDim.x + threadIdx.x;
  long stride = (long)gridDim.x * blockDim.x;
  float4 z = {0.0f, 0.0f, 0.0f, 0.0f};
  for (; i < n4; i += stride) p[i] = z;
}

// ---------------- WMMA helpers --------------------------------------------
__device__ __forceinline__ v8f wmma_f16(v16h a, v16h b, v8f c) {
  // (neg_a, A, neg_b, B, c_mod, C, reuse_a, reuse_b)
  return __builtin_amdgcn_wmma_f32_16x16x32_f16(false, a, false, b,
                                                (short)0, c, false, false);
}

// A-fragment (16x32 f16) from f32 matrix, row-major stride 256.
// lanes 0-15: M=l, K = k0+{0..7, 16..23}; lanes 16-31: M=l, K = k0+{8..15, 24..31}.
__device__ __forceinline__ v16h load_a_x(const float* __restrict__ xch,
                                         int m0, int k0, int lane) {
  int half = lane >> 4, l = lane & 15;
  const float* p = xch + (m0 + l) * 256 + k0 + half * 8;
  float4 q0 = *(const float4*)(p);
  float4 q1 = *(const float4*)(p + 4);
  float4 q2 = *(const float4*)(p + 16);
  float4 q3 = *(const float4*)(p + 20);
  v16h a;
  a[0]=(_Float16)q0.x;  a[1]=(_Float16)q0.y;  a[2]=(_Float16)q0.z;  a[3]=(_Float16)q0.w;
  a[4]=(_Float16)q1.x;  a[5]=(_Float16)q1.y;  a[6]=(_Float16)q1.z;  a[7]=(_Float16)q1.w;
  a[8]=(_Float16)q2.x;  a[9]=(_Float16)q2.y;  a[10]=(_Float16)q2.z; a[11]=(_Float16)q2.w;
  a[12]=(_Float16)q3.x; a[13]=(_Float16)q3.y; a[14]=(_Float16)q3.z; a[15]=(_Float16)q3.w;
  return a;
}

// A-fragment from f16 matrix, row-major stride 256.
__device__ __forceinline__ v16h load_a_h(const _Float16* __restrict__ base,
                                         int m0, int k0, int lane) {
  int half = lane >> 4, l = lane & 15;
  const _Float16* p = base + (m0 + l) * 256 + k0 + half * 8;
  union { v16h v; uint4 u[2]; } r;
  r.u[0] = *(const uint4*)(p);        // 8 halves: K = k0+half*8 .. +7
  r.u[1] = *(const uint4*)(p + 16);   // 8 halves: K = k0+16+half*8 .. +7
  return r.v;
}

// B-fragment (32x16 f16): lane holds one N column, K = k0+half*8+{0..7, 16..23}.
// Source stored transposed: base[col][j], stride 256.
__device__ __forceinline__ v16h load_b_h(const _Float16* __restrict__ base,
                                         int n0, int k0, int lane) {
  int half = lane >> 4, l = lane & 15;
  const _Float16* p = base + (n0 + l) * 256 + k0 + half * 8;
  union { v16h v; uint4 u[2]; } r;
  r.u[0] = *(const uint4*)(p);
  r.u[1] = *(const uint4*)(p + 16);
  return r.v;
}

// B-fragment from LDS staging buffer (rows = stage-1 col, stride LDSTR halves)
__device__ __forceinline__ v16h load_b_lds(const _Float16* lds,
                                           int row0, int k0, int lane) {
  int half = lane >> 4, l = lane & 15;
  const _Float16* p = lds + (row0 + l) * LDSTR + k0 + half * 8;
  union { v16h v; uint4 u[2]; } r;
  r.u[0] = *(const uint4*)(p);
  r.u[1] = *(const uint4*)(p + 16);
  return r.v;
}

// ---------------- fused per-channel kernel --------------------------------
__global__ void __launch_bounds__(256, 1)
pdiae_fused_kernel(const float* __restrict__ x,
                   const _Float16* __restrict__ bt,
                   const _Float16* __restrict__ a8,
                   float* __restrict__ out) {
  extern __shared__ _Float16 lds[];   // 512 * LDSTR halves = 264 KB
  const int ch   = blockIdx.x;        // b*32 + c, 0..255
  const int tid  = threadIdx.x;
  const int wave = tid >> 5;          // 8 wave32 per block
  const int lane = tid & 31;
  const int half = lane >> 4, l = lane & 15;
  const float* xch = x + (long)ch * (256 * 256);

  // ---- Stage 1: [U|V|Uw|Vw] = x * Bt^T, stored transposed in LDS ----------
  // 512 output tiles (16 i-tiles x 32 col-tiles), 64 per wave.
  for (int t = wave; t < 512; t += 8) {
    int mt = t & 15, nt = t >> 4;
    int m0 = mt * 16, n0 = nt * 16;
    v8f acc = {};
    #pragma unroll
    for (int kk = 0; kk < 8; ++kk)
      acc = wmma_f16(load_a_x(xch, m0, kk * 32, lane),
                     load_b_h(bt, n0, kk * 32, lane), acc);
    // C/D layout: vgpr r -> (M = half*8 + r, N = l). Transpose into LDS:
    // row = global col (== var*128 + k2c), contiguous i -> one b128 store.
    int col = n0 + l;
    _Float16* dst = lds + (long)col * LDSTR + m0 + half * 8;
    union { uint4 u; _Float16 h[8]; } pk;
    #pragma unroll
    for (int r = 0; r < 8; ++r) pk.h[r] = (_Float16)acc[r];
    *(uint4*)dst = pk.u;
  }
  __syncthreads();

  // ---- Stage 2: 64x128 complex output tiles, 4 per wave -------------------
  for (int q = 0; q < 4; ++q) {
    int tt  = wave * 4 + q;           // 0..31
    int k1t = tt >> 3, k2t = tt & 7;
    int m0  = k1t * 16;
    v8f acc1 = {}, acc2 = {}, acc3 = {}, acc4 = {};
    for (int kk = 0; kk < 16; ++kk) {
      int i0 = kk * 32;
      v16h aWc  = load_a_h(a8 + 0 * 16384, m0, i0, lane);
      v16h aWs  = load_a_h(a8 + 1 * 16384, m0, i0, lane);
      v16h aWsN = load_a_h(a8 + 2 * 16384, m0, i0, lane);
      v16h aC   = load_a_h(a8 + 3 * 16384, m0, i0, lane);
      v16h aS   = load_a_h(a8 + 4 * 16384, m0, i0, lane);
      v16h aSN  = load_a_h(a8 + 5 * 16384, m0, i0, lane);
      v16h bU   = load_b_lds(lds, 0 * 128 + k2t * 16, i0, lane);
      v16h bV   = load_b_lds(lds, 1 * 128 + k2t * 16, i0, lane);
      v16h bUw  = load_b_lds(lds, 2 * 128 + k2t * 16, i0, lane);
      v16h bVw  = load_b_lds(lds, 3 * 128 + k2t * 16, i0, lane);
      acc1 = wmma_f16(aWc, bU,  acc1);  acc1 = wmma_f16(aWsN, bV,  acc1);
      acc2 = wmma_f16(aC,  bUw, acc2);  acc2 = wmma_f16(aSN,  bVw, acc2);
      acc3 = wmma_f16(aWc, bV,  acc3);  acc3 = wmma_f16(aWs,  bU,  acc3);
      acc4 = wmma_f16(aC,  bVw, acc4);  acc4 = wmma_f16(aS,   bUw, acc4);
    }
    // Epilogue: out = (f1*T1 + f2*T2)/256, im negated (exp(-i...) kernel).
    int k2c = k2t * 16 + l;
    int k2  = (k2c < 64) ? k2c : k2c + 1;   // skip unwritten column 64
    float f2 = k2 * (1.0f / 256.0f);
    float* outc = out + (long)ch * (256 * 129 * 2);
    #pragma unroll
    for (int r = 0; r < 8; ++r) {
      int k1 = m0 + half * 8 + r;
      float f1 = k1 * (1.0f / 256.0f);
      float2 val;
      val.x =  (f1 * acc1[r] + f2 * acc2[r]) * (1.0f / 256.0f);
      val.y = -(f1 * acc3[r] + f2 * acc4[r]) * (1.0f / 256.0f);
      *(float2*)(outc + ((long)k1 * 129 + k2) * 2) = val;
    }
  }
}

// ---------------------------------------------------------------------------
extern "C" void kernel_launch(void* const* d_in, const int* in_sizes, int n_in,
                              void* d_out, int out_size, void* d_ws, size_t ws_size,
                              hipStream_t stream) {
  const float* x = (const float*)d_in[0];
  // d_in[1] = modes (always 64 in this problem); grid structure is hardcoded.
  float* out = (float*)d_out;
  _Float16* bt = (_Float16*)d_ws;         // 512*256 halves (256 KB)
  _Float16* a8 = bt + NB_HALVES;          // 6*64*256 halves (192 KB)

  // 1) build DFT basis (229376 threads exactly)
  basis_init_kernel<<<896, 256, 0, stream>>>(bt, a8);

  // 2) zero the full complex output (8*32*256*129 complex64 = 16.9M floats)
  long n_floats = (long)256 * 256 * 129 * 2;
  zero_out_kernel<<<2048, 256, 0, stream>>>((float4*)out, n_floats / 4);

  // 3) fused two-stage WMMA DFT, one workgroup per channel
  size_t lds_bytes = (size_t)512 * LDSTR * sizeof(_Float16);  // 264 KB
  pdiae_fused_kernel<<<256, 256, lds_bytes, stream>>>(x, bt, a8, out);
}
